// GEARS_Model_42554535969482
// MI455X (gfx1250) — compile-verified, matrix operand order checked by reading, not run on hardware
//
#include <hip/hip_runtime.h>

// ---------------- constants ----------------
namespace {
constexpr int   kB = 64;
constexpr int   kG = 3040;
constexpr int   kH = 64;
constexpr int   kP = 5000;
constexpr long  kN = (long)kB * kG;          // 194560
constexpr int   kE1 = 97280;
constexpr int   kE2 = 160000;
constexpr float kEPS = 1e-5f;
}

typedef __attribute__((ext_vector_type(16))) __bf16 v16bf;
typedef __attribute__((ext_vector_type(8)))  float  v8f;

// fp32 -> bf16 round-to-nearest-even, packed pair
__device__ __forceinline__ unsigned f2bf_bits(float f) {
  unsigned u = __float_as_uint(f);
  return (u + 0x7FFFu + ((u >> 16) & 1u)) >> 16;
}
__device__ __forceinline__ unsigned pack2(float lo, float hi) {
  return f2bf_bits(lo) | (f2bf_bits(hi) << 16);
}

// ---------------- small utility kernels ----------------
__global__ void k_fill(float* __restrict__ p, float v, long n) {
  long i = (long)blockIdx.x * blockDim.x + threadIdx.x;
  if (i < n) p[i] = v;
}

__global__ void k_copy(const float* __restrict__ a, float* __restrict__ b, long n) {
  long i = (long)blockIdx.x * blockDim.x + threadIdx.x;
  if (i < n) b[i] = a[i];
}

// row renorm to max-norm 1 (torch Embedding max_norm=1 semantics), H=64
__global__ __launch_bounds__(64)
void k_renorm(const float* __restrict__ in, float* __restrict__ out, int rows) {
  __shared__ float sh[64];
  int r = blockIdx.x;
  if (r >= rows) return;
  int t = threadIdx.x;
  float v = in[(size_t)r * kH + t];
  sh[t] = v * v;
  __syncthreads();
  for (int off = 32; off > 0; off >>= 1) {
    if (t < off) sh[t] += sh[t + off];
    __syncthreads();
  }
  float nrm = sqrtf(sh[0]);
  float scale = (nrm > 1.f) ? 1.f / nrm : 1.f;
  out[(size_t)r * kH + t] = v * scale;
}

// deg[col[e]] += w[e]
__global__ void k_deg(const int* __restrict__ cols_, const float* __restrict__ w,
                      float* __restrict__ deg, int E) {
  int e = blockIdx.x * blockDim.x + threadIdx.x;
  if (e < E) atomicAdd(&deg[cols_[e]], w[e]);
}

// self loop: agg[i,h] = x[i,h] / deg[i]
__global__ void k_selfloop(const float* __restrict__ x, const float* __restrict__ deg,
                           float* __restrict__ agg, long n64) {
  long i = (long)blockIdx.x * blockDim.x + threadIdx.x;
  if (i >= n64) return;
  long r = i >> 6;
  agg[i] = x[i] / deg[r];
}

// edge scatter: agg[col,h] += w * rsqrt(deg[row]) * rsqrt(deg[col]) * x[row,h]
__global__ __launch_bounds__(256)
void k_agg_edges(const int* __restrict__ rows_, const int* __restrict__ cols_,
                 const float* __restrict__ w, const float* __restrict__ deg,
                 const float* __restrict__ x, float* __restrict__ agg, int E) {
  int e = blockIdx.x * 4 + (threadIdx.x >> 6);
  int h = threadIdx.x & 63;
  if (e >= E) return;
  int rs = rows_[e], cd = cols_[e];
  float coef = w[e] * rsqrtf(deg[rs]) * rsqrtf(deg[cd]);
  atomicAdd(&agg[(size_t)cd * kH + h], coef * x[(size_t)rs * kH + h]);
}

// ---------------- WMMA GEMM: C[M,NT] = A[M,K] @ W[K,NT] + bias ----------------
// bf16 operands, f32 accumulate. Tiles are stored in LDS pre-swizzled into
// WMMA fragment order: element k (0..31) of a row maps to lane-group
// lg = (k>>3)&1, fragment slot e = ((k>>4)<<3)|(k&7).  Each lane then loads
// its entire 16-element fragment as one contiguous 32B v16bf (2x ds_load_b128).
// Full blocks (row0+128 <= M, the dominant case) take a uniform unguarded
// fast path: straight global_load_b64 stream, no exec-mask churn.
template <int NT>
__global__ __launch_bounds__(256)
void k_gemm(const float* __restrict__ A, const float* __restrict__ Wm,
            const float* __restrict__ bias, float* __restrict__ C,
            int M, int K) {
  __shared__ unsigned lA[128 * 16];   // 8 KB: [(r*2+lg)*8 + e/2] packed bf16 pairs
  __shared__ unsigned lB[16 * NT];    // [(n*2+lg)*8 + e/2]
  const int t    = threadIdx.x;
  const int wave = t >> 5;
  const int lane = t & 31;
  const int row0 = blockIdx.x * 128;
  const bool full = (row0 + 128 <= M);   // uniform across block (scalar branch)

  v8f acc[NT / 16];
#pragma unroll
  for (int i = 0; i < NT / 16; ++i)
#pragma unroll
    for (int e = 0; e < 8; ++e) acc[i][e] = 0.f;

  for (int k0 = 0; k0 < K; k0 += 32) {
    // ---- A tile fill: 128 rows x 16 k-pairs = 2048 u32 stores, swizzled ----
    if (full) {
#pragma unroll
      for (int i = 0; i < 8; ++i) {
        int p = t + i * 256;         // 0..2047
        int r = p >> 4;              // 0..127
        int k = (p & 15) << 1;       // even k
        const float* ap = A + (size_t)(row0 + r) * K + k0 + k;
        float v0 = ap[0];
        float v1 = ap[1];
        int lg = (k >> 3) & 1;
        int e  = ((k >> 4) << 3) | (k & 7);
        lA[(r * 2 + lg) * 8 + (e >> 1)] = pack2(v0, v1);
      }
    } else {
#pragma unroll
      for (int i = 0; i < 8; ++i) {
        int p  = t + i * 256;
        int r  = p >> 4;
        int k  = (p & 15) << 1;
        int gr = row0 + r;
        float v0 = 0.f, v1 = 0.f;
        if (gr < M) {
          const float* ap = A + (size_t)gr * K + k0 + k;
          v0 = ap[0];
          v1 = ap[1];
        }
        int lg = (k >> 3) & 1;
        int e  = ((k >> 4) << 3) | (k & 7);
        lA[(r * 2 + lg) * 8 + (e >> 1)] = pack2(v0, v1);
      }
    }
    // ---- B tile fill: 16 k-pairs x NT cols, swizzled (always in-bounds) ----
#pragma unroll
    for (int i = 0; i < (16 * NT) / 256; ++i) {
      int p = t + i * 256;
      int k = (p / NT) << 1;
      int n = p % NT;
      float v0 = Wm[(size_t)(k0 + k) * NT + n];
      float v1 = Wm[(size_t)(k0 + k + 1) * NT + n];
      int lg = (k >> 3) & 1;
      int e  = ((k >> 4) << 3) | (k & 7);
      lB[(n * 2 + lg) * 8 + (e >> 1)] = pack2(v0, v1);
    }
    // prefetch next A k-slice while we compute (lowers to global_prefetch)
    if (k0 + 32 < K && full) {
      int r = (t >> 1) & 127;
      __builtin_prefetch(&A[(size_t)(row0 + r) * K + k0 + 32], 0, 0);
    }
    __syncthreads();

    const int mloc = wave * 16 + (lane & 15);
    const int lg   = lane >> 4;
    v16bf afrag = *reinterpret_cast<const v16bf*>(&lA[(mloc * 2 + lg) * 8]);
#pragma unroll
    for (int nt = 0; nt < NT / 16; ++nt) {
      int nloc = nt * 16 + (lane & 15);
      v16bf bfrag = *reinterpret_cast<const v16bf*>(&lB[(nloc * 2 + lg) * 8]);
      acc[nt] = __builtin_amdgcn_wmma_f32_16x16x32_bf16(
          false, afrag, false, bfrag, (short)0, acc[nt], false, false);
    }
    __syncthreads();
  }

  // D layout: VGPR r -> M = r + 8*(lane/16), N = lane%16
  const int mbase = row0 + wave * 16;
  const int nlane = lane & 15;
  const int moff  = (lane >> 4) * 8;
  if (full) {
#pragma unroll
    for (int nt = 0; nt < NT / 16; ++nt) {
#pragma unroll
      for (int r = 0; r < 8; ++r) {
        int m = mbase + moff + r;
        int n = nt * 16 + nlane;
        C[(size_t)m * NT + n] = acc[nt][r] + bias[n];
      }
    }
  } else {
#pragma unroll
    for (int nt = 0; nt < NT / 16; ++nt) {
#pragma unroll
      for (int r = 0; r < 8; ++r) {
        int m = mbase + moff + r;
        int n = nt * 16 + nlane;
        if (m < M) C[(size_t)m * NT + n] = acc[nt][r] + bias[n];
      }
    }
  }
}

// ---------------- BatchNorm building blocks ----------------
// weighted column sums/sumsq (rows < wb get w0, else w1); atomic accumulate
__global__ __launch_bounds__(256)
void k_colstats(const float* __restrict__ X, float* __restrict__ sums,
                float* __restrict__ sumsq, long rows, int cols,
                long wb, float w0, float w1) {
  int t = threadIdx.x;
  int col = t % cols;
  int rl  = t / cols;
  int rlanes = 256 / cols;
  float s = 0.f, sq = 0.f;
  for (long r = (long)blockIdx.x * rlanes + rl; r < rows; r += (long)gridDim.x * rlanes) {
    float v = X[r * cols + col];
    float w = (r < wb) ? w0 : w1;
    s += w * v;
    sq += w * v * v;
  }
  atomicAdd(&sums[col], s);
  atomicAdd(&sumsq[col], sq);
}

__global__ void k_finalize(const float* __restrict__ sums, const float* __restrict__ sumsq,
                           float* __restrict__ mu, float* __restrict__ rstd,
                           int cols, float invcount) {
  int c = blockIdx.x * blockDim.x + threadIdx.x;
  if (c < cols) {
    float m = sums[c] * invcount;
    float v = sumsq[c] * invcount - m * m;
    mu[c] = m;
    rstd[c] = rsqrtf(v + kEPS);
  }
}

__global__ void k_bn_apply(const float* __restrict__ in, float* __restrict__ out,
                           const float* __restrict__ mu, const float* __restrict__ rstd,
                           const float* __restrict__ g, const float* __restrict__ be,
                           long n, int cols, int relu) {
  long i = (long)blockIdx.x * blockDim.x + threadIdx.x;
  if (i >= n) return;
  int c = (int)(i % cols);
  float v = (in[i] - mu[c]) * rstd[c] * g[c] + be[c];
  if (relu) v = fmaxf(v, 0.f);
  out[i] = v;
}

// ---------------- model-specific fusion kernels ----------------
// stk[r,h] = base_g[r%G,h] + 0.2*pos2[r,h],  r in [0,2G)
__global__ void k_stk(const float* __restrict__ base_g, const float* __restrict__ pos2,
                      float* __restrict__ out, long n) {
  long i = (long)blockIdx.x * blockDim.x + threadIdx.x;
  if (i >= n) return;
  long r = i >> 6;
  int  h = (int)(i & 63);
  long g = r % kG;
  out[i] = base_g[g * kH + h] + 0.2f * pos2[i];
}

// pert_sum[b,h] = pge[idx[b,0],h] + pge[idx[b,1],h]
__global__ void k_pert_sum(const float* __restrict__ pge, const int* __restrict__ idx,
                           float* __restrict__ out) {
  int i = blockIdx.x * blockDim.x + threadIdx.x;  // B*64
  if (i >= kB * kH) return;
  int b = i >> 6, h = i & 63;
  out[i] = pge[(size_t)idx[b * 2] * kH + h] + pge[(size_t)idx[b * 2 + 1] * kH + h];
}

// base_pb[b*G+g,h] = etv2o[(b>0?G:0)+g,h] + et[b,h]
__global__ void k_base_pb(const float* __restrict__ etv2o, const float* __restrict__ et,
                          float* __restrict__ out, long n) {
  long i = (long)blockIdx.x * blockDim.x + threadIdx.x;
  if (i >= n) return;
  long row = i >> 6;
  int  h   = (int)(i & 63);
  long b = row / kG;
  long g = row % kG;
  long src = ((b > 0) ? (long)kG : 0) + g;
  out[i] = etv2o[src * kH + h] + et[b * kH + h];
}

// o1[row] = dot64(out[row,:], w1[row%G,:]) + b1[row%G]  (one wave per row)
__global__ __launch_bounds__(256)
void k_o1(const float* __restrict__ X, const float* __restrict__ w1,
          const float* __restrict__ b1, float* __restrict__ o1, long rows) {
  int wave = threadIdx.x >> 5;
  int lane = threadIdx.x & 31;
  long row = (long)blockIdx.x * 8 + wave;
  if (row >= rows) return;
  long g = row % kG;
  const float* xr = X + row * kH;
  const float* wr = w1 + g * kH;
  float s = xr[lane] * wr[lane] + xr[lane + 32] * wr[lane + 32];
  for (int off = 16; off > 0; off >>= 1) s += __shfl_down(s, off, 32);
  if (lane == 0) o1[row] = s + b1[g];
}

// res[b,g] = o1[b,g]*w2[g,0] + dot64(cg[b,:], w2[g,1:65]) + b2[g] + x[b*G+g]
__global__ __launch_bounds__(256)
void k_res(const float* __restrict__ o1, const float* __restrict__ cg,
           const float* __restrict__ w2, const float* __restrict__ b2,
           const float* __restrict__ x, float* __restrict__ out, long n) {
  __shared__ float scg[kB * kH];  // 16 KB: full cross-gene state
  for (int i = threadIdx.x; i < kB * kH; i += 256) scg[i] = cg[i];
  __syncthreads();
  long i = (long)blockIdx.x * 256 + threadIdx.x;
  if (i >= n) return;
  long b = i / kG;
  long g = i % kG;
  const float* wr = w2 + g * (kH + 1);
  const float* cr = scg + b * kH;
  float s = o1[i] * wr[0] + b2[g] + x[i];
#pragma unroll
  for (int k = 0; k < kH; ++k) s += cr[k] * wr[1 + k];
  out[i] = s;
}

// ---------------- launch ----------------
static inline unsigned cdiv(long n, long d) { return (unsigned)((n + d - 1) / d); }

extern "C" void kernel_launch(void* const* d_in, const int* in_sizes, int n_in,
                              void* d_out, int out_size, void* d_ws, size_t ws_size,
                              hipStream_t stream) {
  // -------- inputs --------
  const float* x        = (const float*)d_in[0];
  const int*   pert_idx = (const int*)d_in[1];
  const int*   e_co     = (const int*)d_in[2];   // [2,E1]: rows, cols
  const float* w_co     = (const float*)d_in[3];
  const int*   e_go     = (const int*)d_in[4];
  const float* w_go     = (const float*)d_in[5];
  int i6 = 6;
  const float* gene_emb = (const float*)d_in[i6++];
  const float* emb_pos  = (const float*)d_in[i6++];
  const float* pert_emb = (const float*)d_in[i6++];
  const float* sg_co_W  = (const float*)d_in[i6++];
  const float* sg_co_b  = (const float*)d_in[i6++];
  const float* sg_go_W  = (const float*)d_in[i6++];
  const float* sg_go_b  = (const float*)d_in[i6++];
  const float* bn_emb_g = (const float*)d_in[i6++];
  const float* bn_emb_be= (const float*)d_in[i6++];
  const float* bn_pb_g  = (const float*)d_in[i6++];
  const float* bn_pb_be = (const float*)d_in[i6++];
  // mlps: W1,b1,g1,be1,W2,b2,g2,be2
  const float* etv2[8]; for (int i=0;i<8;++i) etv2[i]=(const float*)d_in[i6++];
  const float* pf[8];   for (int i=0;i<8;++i) pf[i]  =(const float*)d_in[i6++];
  const float* rw[8];   for (int i=0;i<8;++i) rw[i]  =(const float*)d_in[i6++];
  const float* cgp[8];  for (int i=0;i<8;++i) cgp[i] =(const float*)d_in[i6++];
  const float* indv_w1 = (const float*)d_in[i6++];
  const float* indv_b1 = (const float*)d_in[i6++];
  const float* indv_w2 = (const float*)d_in[i6++];
  const float* indv_b2 = (const float*)d_in[i6++];
  float* out = (float*)d_out;
  (void)in_sizes; (void)n_in; (void)out_size;

  // -------- workspace layout (floats) --------
  float* Wp = (float*)d_ws;
  size_t off = 0;
  auto alloc = [&](size_t n) { float* p = Wp + off; off += n; return p; };
  float* rn_gene = alloc((size_t)kG * kH);
  float* rn_pos  = alloc((size_t)kG * kH);
  float* rn_pert = alloc((size_t)kP * kH);
  float* deg_co  = alloc(kG);
  float* deg_go  = alloc(kP);
  float* agg2    = alloc((size_t)2 * kG * kH);   // co agg (rows<G) + plain (rows>=G); reused as stk
  float* pos2    = alloc((size_t)2 * kG * kH);
  float* agg_go  = alloc((size_t)kP * kH);
  float* pge     = alloc((size_t)kP * kH);
  float* base_g  = alloc((size_t)kG * kH);
  float* et_h    = alloc((size_t)2 * kG * kH);
  float* et_o    = alloc((size_t)2 * kG * kH);
  float* psum    = alloc((size_t)kB * kH);
  float* pf_h    = alloc((size_t)kB * kH);
  float* pf_o    = alloc((size_t)kB * kH);
  float* st      = alloc(512);                   // sums|sumsq|mu|rstd (128 each)
  float* bigA    = alloc((size_t)kN * kH);       // base_pb, later rw output
  float* h_rw    = alloc((size_t)kN * 2 * kH);
  float* o1v     = alloc((size_t)kN);
  float* cg_h    = alloc((size_t)kB * kH);
  float* cg_o    = alloc((size_t)kB * kH);
  if (off * sizeof(float) > ws_size) return;     // insufficient scratch

  float* sums = st, *sumsq = st + 128, *mu = st + 256, *rstd = st + 384;

  auto bn = [&](const float* in, float* o, const float* g, const float* be,
                long rows, int cols, long wb, float w0, float w1,
                float invcount, int relu) {
    k_fill<<<1, 256, 0, stream>>>(st, 0.f, 256);
    k_colstats<<<512, 256, 0, stream>>>(in, sums, sumsq, rows, cols, wb, w0, w1);
    k_finalize<<<1, 128, 0, stream>>>(sums, sumsq, mu, rstd, cols, invcount);
    long n = rows * cols;
    k_bn_apply<<<cdiv(n, 256), 256, 0, stream>>>(in, o, mu, rstd, g, be, n, cols, relu);
  };

  // -------- 1. renorm embeddings --------
  k_renorm<<<kG, 64, 0, stream>>>(gene_emb, rn_gene, kG);
  k_renorm<<<kG, 64, 0, stream>>>(emb_pos, rn_pos, kG);
  k_renorm<<<kP, 64, 0, stream>>>(pert_emb, rn_pert, kP);

  // -------- 2. base = relu(BN(emb)); tiled rows => stats over G --------
  bn(rn_gene, base_g, bn_emb_g, bn_emb_be, kG, kH, kG, 1.f, 1.f, 1.f / kG, 1);

  // -------- 3. coexpression SGConv (2-variant: conv rows + plain rows) --------
  k_fill<<<cdiv(kG, 256), 256, 0, stream>>>(deg_co, 1.f, kG);
  k_deg<<<cdiv(kE1, 256), 256, 0, stream>>>(e_co + kE1, w_co, deg_co, kE1);
  k_selfloop<<<cdiv((long)kG * kH, 256), 256, 0, stream>>>(rn_pos, deg_co, agg2, (long)kG * kH);
  k_agg_edges<<<cdiv(kE1, 4), 256, 0, stream>>>(e_co, e_co + kE1, w_co, deg_co, rn_pos, agg2, kE1);
  k_copy<<<cdiv((long)kG * kH, 256), 256, 0, stream>>>(rn_pos, agg2 + (size_t)kG * kH, (long)kG * kH);
  k_gemm<64><<<cdiv(2 * kG, 128), 256, 0, stream>>>(agg2, sg_co_W, sg_co_b, pos2, 2 * kG, kH);

  // -------- 4. etv2 MLP on 2G distinct rows, weighted BN (1 vs 63) --------
  k_stk<<<cdiv((long)2 * kG * kH, 256), 256, 0, stream>>>(base_g, pos2, agg2, (long)2 * kG * kH);
  k_gemm<64><<<cdiv(2 * kG, 128), 256, 0, stream>>>(agg2, etv2[0], etv2[1], et_h, 2 * kG, kH);
  bn(et_h, et_h, etv2[2], etv2[3], 2 * kG, kH, kG, 1.f, 63.f, 1.f / (float)kN, 1);
  k_gemm<64><<<cdiv(2 * kG, 128), 256, 0, stream>>>(et_h, etv2[4], etv2[5], et_o, 2 * kG, kH);
  bn(et_o, et_o, etv2[6], etv2[7], 2 * kG, kH, kG, 1.f, 63.f, 1.f / (float)kN, 0);

  // -------- 5. GO SGConv + pert fuse MLP --------
  k_fill<<<cdiv(kP, 256), 256, 0, stream>>>(deg_go, 1.f, kP);
  k_deg<<<cdiv(kE2, 256), 256, 0, stream>>>(e_go + kE2, w_go, deg_go, kE2);
  k_selfloop<<<cdiv((long)kP * kH, 256), 256, 0, stream>>>(rn_pert, deg_go, agg_go, (long)kP * kH);
  k_agg_edges<<<cdiv(kE2, 4), 256, 0, stream>>>(e_go, e_go + kE2, w_go, deg_go, rn_pert, agg_go, kE2);
  k_gemm<64><<<cdiv(kP, 128), 256, 0, stream>>>(agg_go, sg_go_W, sg_go_b, pge, kP, kH);
  k_pert_sum<<<cdiv(kB * kH, 256), 256, 0, stream>>>(pge, pert_idx, psum);
  k_gemm<64><<<1, 256, 0, stream>>>(psum, pf[0], pf[1], pf_h, kB, kH);
  bn(pf_h, pf_h, pf[2], pf[3], kB, kH, kB, 1.f, 1.f, 1.f / kB, 1);
  k_gemm<64><<<1, 256, 0, stream>>>(pf_h, pf[4], pf[5], pf_o, kB, kH);
  bn(pf_o, pf_o, pf[6], pf[7], kB, kH, kB, 1.f, 1.f, 1.f / kB, 0);

  // -------- 6. broadcast add + BN + relu over full N rows --------
  k_base_pb<<<cdiv(kN * kH, 256), 256, 0, stream>>>(et_o, pf_o, bigA, kN * kH);
  bn(bigA, bigA, bn_pb_g, bn_pb_be, kN, kH, kN, 1.f, 1.f, 1.f / (float)kN, 1);

  // -------- 7. recovery_w MLP: N x 64 -> 128 -> 64 --------
  k_gemm<128><<<cdiv(kN, 128), 256, 0, stream>>>(bigA, rw[0], rw[1], h_rw, (int)kN, kH);
  bn(h_rw, h_rw, rw[2], rw[3], kN, 2 * kH, kN, 1.f, 1.f, 1.f / (float)kN, 1);
  k_gemm<64><<<cdiv(kN, 128), 256, 0, stream>>>(h_rw, rw[4], rw[5], bigA, (int)kN, 2 * kH);
  bn(bigA, bigA, rw[6], rw[7], kN, kH, kN, 1.f, 1.f, 1.f / (float)kN, 0);

  // -------- 8. per-gene dot (indv_w1) --------
  k_o1<<<cdiv(kN, 8), 256, 0, stream>>>(bigA, indv_w1, indv_b1, o1v, kN);

  // -------- 9. cross_gene_state MLP: [B,G] @ [G,64] --------
  k_gemm<64><<<1, 256, 0, stream>>>(o1v, cgp[0], cgp[1], cg_h, kB, kG);
  bn(cg_h, cg_h, cgp[2], cgp[3], kB, kH, kB, 1.f, 1.f, 1.f / kB, 1);
  k_gemm<64><<<1, 256, 0, stream>>>(cg_h, cgp[4], cgp[5], cg_o, kB, kH);
  bn(cg_o, cg_o, cgp[6], cgp[7], kB, kH, kB, 1.f, 1.f, 1.f / kB, 0);

  // -------- 10. final combine + residual --------
  k_res<<<cdiv(kN, 256), 256, 0, stream>>>(o1v, cg_o, indv_w2, indv_b2, x, out, kN);
}
